// Gemma4TextAttention_49331994362296
// MI455X (gfx1250) — matrix operand order repaired and checked by reading, not compiled, not run on hardware
//
#include <hip/hip_runtime.h>
#include <hip/hip_bf16.h>

typedef __bf16 bf16;
typedef __attribute__((ext_vector_type(16))) __bf16 v16bf;
typedef __attribute__((ext_vector_type(8)))  float  v8f;

#define S_LEN   4096
#define H_DIM   2048
#define HQ_N    8
#define HKV_N   4
#define D_HEAD  256
#define WIN_SZ  1024
#define SOFTCAP 50.0f
#define EPS_RMS 1e-6f

// ---------------- WMMA fragment builders (contiguous k order: k = 16*(lane/16) + elem) ----

// A-matrix 16x32 bf16: rowp -> &lds[m][0] (32 contiguous K values for row m).
// lanes 0-15: elems 0-15 = K0..15
// lanes 16-31: elems 0-15 = K16..31
__device__ inline v16bf frag_a_rowmajor(const bf16* rowp, int lane) {
  const int kb = (lane >= 16) ? 16 : 0;
  v16bf a;
#pragma unroll
  for (int i = 0; i < 16; ++i) {
    a[i]     = rowp[kb + i];
  }
  return a;
}

// B-matrix 32x16 bf16 when LDS holds it n-major: lds[n][k], row stride rs.
__device__ inline v16bf frag_b_nmajor(const bf16* base, int rs, int lane) {
  const int n    = lane & 15;
  const int koff = (lane >= 16) ? 16 : 0;
  const bf16* p  = base + (size_t)n * rs + koff;
  v16bf b;
#pragma unroll
  for (int i = 0; i < 16; ++i) b[i] = p[i];
  return b;
}

// B-matrix 32x16 bf16 when LDS holds it k-major: lds[k][n], row stride rs.
__device__ inline v16bf frag_b_kmajor(const bf16* base, int rs, int lane) {
  const int n    = lane & 15;
  const int koff = (lane >= 16) ? 16 : 0;
  const bf16* p  = base + (size_t)koff * rs + n;
  v16bf b;
#pragma unroll
  for (int i = 0; i < 16; ++i) b[i] = p[(size_t)i * rs];
  return b;
}

// ================= Kernel 1: QKV projection + RMSNorm (+RoPE) ==============
// grid = (S/32, HQ + 2*HKV); block = 256 (8 waves). Each block owns a full
// 32 x 256 (rows x full head dim) tile: 2x2 WMMA subtiles per wave per K-step.
__global__ __launch_bounds__(256)
void qkv_proj_norm_rope_kernel(const float* __restrict__ hidden,
                               const float* __restrict__ cosb,
                               const float* __restrict__ sinb,
                               const float* __restrict__ Wq,
                               const float* __restrict__ Wk,
                               const float* __restrict__ Wv,
                               const float* __restrict__ qnw,
                               const float* __restrict__ knw,
                               bf16* __restrict__ Qbuf,
                               bf16* __restrict__ Kbuf,
                               bf16* __restrict__ Vbuf) {
  const int s0  = blockIdx.x * 32;
  const int hid = blockIdx.y;      // 0..7 q, 8..11 k, 12..15 v

  const float* W; bf16* Obuf; const float* nw; int mode; int head;
  if (hid < HQ_N)            { W = Wq; Obuf = Qbuf; nw = qnw;     mode = 0; head = hid; }
  else if (hid < HQ_N+HKV_N) { W = Wk; Obuf = Kbuf; nw = knw;     mode = 1; head = hid - HQ_N; }
  else                       { W = Wv; Obuf = Vbuf; nw = nullptr; mode = 2; head = hid - HQ_N - HKV_N; }
  const int wrow0 = head * D_HEAD;

  __shared__ __align__(16) bf16  ldsA[32][32];
  __shared__ __align__(16) bf16  ldsB[256][32];   // n-major: [outcol][k]
  __shared__ __align__(16) float ldsO[32][D_HEAD];
  __shared__ float partS[32][8];
  __shared__ float rrms[32];

  const int t = threadIdx.x, lane = t & 31, wave = t >> 5;

  v8f acc00 = {}, acc01 = {}, acc10 = {}, acc11 = {};

  for (int k0 = 0; k0 < H_DIM; k0 += 32) {
    { // A: 32x32 f32 -> bf16
      const int r = t >> 3, c = (t & 7) * 4;
      float4 v = *(const float4*)(hidden + (size_t)(s0 + r) * H_DIM + k0 + c);
      ldsA[r][c]   = (bf16)v.x; ldsA[r][c+1] = (bf16)v.y;
      ldsA[r][c+2] = (bf16)v.z; ldsA[r][c+3] = (bf16)v.w;
    }
    { // B: 256 rows of W, 32 k (f32 -> bf16)
      const float* wr = W + (size_t)(wrow0 + t) * H_DIM + k0;
#pragma unroll
      for (int c = 0; c < 32; c += 4) {
        float4 v = *(const float4*)(wr + c);
        ldsB[t][c]   = (bf16)v.x; ldsB[t][c+1] = (bf16)v.y;
        ldsB[t][c+2] = (bf16)v.z; ldsB[t][c+3] = (bf16)v.w;
      }
    }
    if (k0 + 32 < H_DIM) {
      __builtin_prefetch(hidden + (size_t)s0 * H_DIM + k0 + 32, 0, 1);
      __builtin_prefetch(W + (size_t)wrow0 * H_DIM + k0 + 32, 0, 1);
    }
    __syncthreads();
    {
      v16bf a0 = frag_a_rowmajor(&ldsA[lane & 15][0],        lane);
      v16bf a1 = frag_a_rowmajor(&ldsA[16 + (lane & 15)][0], lane);
      v16bf b0 = frag_b_nmajor(&ldsB[wave * 32][0],      32, lane);
      v16bf b1 = frag_b_nmajor(&ldsB[wave * 32 + 16][0], 32, lane);
      acc00 = __builtin_amdgcn_wmma_f32_16x16x32_bf16(false, a0, false, b0, (short)0, acc00, false, false);
      acc01 = __builtin_amdgcn_wmma_f32_16x16x32_bf16(false, a0, false, b1, (short)0, acc01, false, false);
      acc10 = __builtin_amdgcn_wmma_f32_16x16x32_bf16(false, a1, false, b0, (short)0, acc10, false, false);
      acc11 = __builtin_amdgcn_wmma_f32_16x16x32_bf16(false, a1, false, b1, (short)0, acc11, false, false);
    }
    __syncthreads();
  }

  { // spill accumulators to LDS (C layout: vgpr r -> row r(+8), lane&15 -> col)
    const int mo = (lane >= 16) ? 8 : 0;
    const int n0 = wave * 32 + (lane & 15);
#pragma unroll
    for (int r = 0; r < 8; ++r) {
      ldsO[mo + r][n0]           = acc00[r];
      ldsO[mo + r][n0 + 16]      = acc01[r];
      ldsO[16 + mo + r][n0]      = acc10[r];
      ldsO[16 + mo + r][n0 + 16] = acc11[r];
    }
  }
  __syncthreads();

  { // RMS partial sums of squares (row = t>>3, 32 cols per thread)
    const int row = t >> 3, seg = t & 7;
    float s = 0.f;
#pragma unroll
    for (int j = 0; j < 32; ++j) { float x = ldsO[row][seg * 32 + j]; s += x * x; }
    partS[row][seg] = s;
  }
  __syncthreads();
  if (t < 32) {
    float s = 0.f;
#pragma unroll
    for (int j = 0; j < 8; ++j) s += partS[t][j];
    rrms[t] = rsqrtf(s * (1.0f / D_HEAD) + EPS_RMS);
  }
  __syncthreads();

  { // normalize (+weight) in place
    const int row = t >> 3, seg = t & 7;
    const float rm = rrms[row];
#pragma unroll
    for (int j = 0; j < 32; ++j) {
      const int n = seg * 32 + j;
      float y = ldsO[row][n] * rm;
      if (mode != 2) y *= nw[n];
      ldsO[row][n] = y;
    }
  }
  __syncthreads();

  { // RoPE (q/k) and store bf16, head-major [head][s][d]
    const int row = t >> 3, seg = t & 7;
    const int sg  = s0 + row;
    bf16* orow = Obuf + ((size_t)head * S_LEN + sg) * D_HEAD;
#pragma unroll
    for (int j = 0; j < 32; ++j) {
      const int n = seg * 32 + j;
      float y = ldsO[row][n];
      float o;
      if (mode == 2) {
        o = y;
      } else {
        const float c  = cosb[(size_t)sg * D_HEAD + n];
        const float si = sinb[(size_t)sg * D_HEAD + n];
        const float rot = (n < D_HEAD / 2) ? -ldsO[row][n + D_HEAD / 2]
                                           :  ldsO[row][n - D_HEAD / 2];
        o = y * c + rot * si;
      }
      orow[n] = (bf16)o;
    }
  }
}

// ================= Kernel 2: sliding-window flash attention =================
// grid = (S/32, HQ); block = 256. 32 queries x full D=256, 32-key tiles.
__global__ __launch_bounds__(256)
void flash_attn_kernel(const bf16* __restrict__ Qbuf,
                       const bf16* __restrict__ Kbuf,
                       const bf16* __restrict__ Vbuf,
                       bf16* __restrict__ Ctx) {
  const int q0  = blockIdx.x * 32;
  const int h   = blockIdx.y;
  const int kvh = h >> 1;                 // GQA: n_rep = 2
  const int t = threadIdx.x, lane = t & 31, wave = t >> 5;

  __shared__ __align__(16) bf16  ldsQ[32][D_HEAD];
  __shared__ __align__(16) bf16  ldsK[32][D_HEAD];
  __shared__ __align__(16) bf16  ldsV[32][D_HEAD];
  __shared__ float ldsS[32][32];
  __shared__ __align__(16) bf16  ldsP[32][32];
  __shared__ float partS[32][8];
  __shared__ float rowM[32], rowL[32], rowScale[32];

  { // load Q tile 32x256 bf16
    const int r = t >> 3, cb = (t & 7) * 32;
    const uint4* src = (const uint4*)(Qbuf + ((size_t)h * S_LEN + q0 + r) * D_HEAD + cb);
    uint4* dst = (uint4*)(&ldsQ[r][cb]);
#pragma unroll
    for (int i = 0; i < 4; ++i) dst[i] = src[i];
  }
  if (t < 32) { rowM[t] = -3.0e38f; rowL[t] = 0.f; }

  v8f o00 = {}, o01 = {}, o10 = {}, o11 = {};   // [m-subtile][d-subtile]
  int kbeg = q0 - (WIN_SZ - 1); if (kbeg < 0) kbeg = 0; kbeg &= ~31;
  __syncthreads();

  for (int kt = kbeg; kt <= q0 + 31; kt += 32) {
    { // load K,V tile 32x256 (kt+31 <= q0+31 <= S-1, never OOB)
      const int r = t >> 3, cb = (t & 7) * 32;
      const uint4* sk = (const uint4*)(Kbuf + ((size_t)kvh * S_LEN + kt + r) * D_HEAD + cb);
      const uint4* sv = (const uint4*)(Vbuf + ((size_t)kvh * S_LEN + kt + r) * D_HEAD + cb);
      uint4* dk = (uint4*)(&ldsK[r][cb]);
      uint4* dv = (uint4*)(&ldsV[r][cb]);
#pragma unroll
      for (int i = 0; i < 4; ++i) { dk[i] = sk[i]; dv[i] = sv[i]; }
    }
    __syncthreads();

    if (wave < 4) { // S = Q K^T : 32x32 scores = 4 tiles across waves 0..3
      const int mt = wave >> 1, nt = wave & 1;
      v8f sacc = {};
#pragma unroll
      for (int ks = 0; ks < D_HEAD; ks += 32) {
        v16bf a = frag_a_rowmajor(&ldsQ[mt * 16 + (lane & 15)][ks], lane);
        v16bf b = frag_b_nmajor(&ldsK[nt * 16][ks], D_HEAD, lane);
        sacc = __builtin_amdgcn_wmma_f32_16x16x32_bf16(false, a, false, b, (short)0, sacc, false, false);
      }
      const int mo = (lane >= 16) ? 8 : 0;
      const int n  = nt * 16 + (lane & 15);
#pragma unroll
      for (int r = 0; r < 8; ++r) ldsS[mt * 16 + mo + r][n] = sacc[r];
    }
    __syncthreads();

    { // softcap + analytic sliding-window causal mask (4 elems/thread)
      const int row = t >> 3, c0 = (t & 7) * 4;
      const int qg = q0 + row;
#pragma unroll
      for (int e = 0; e < 4; ++e) {
        const int c = c0 + e;
        const int diff = qg - (kt + c);
        float sv = ldsS[row][c];
        sv = tanhf(sv * (1.0f / SOFTCAP)) * SOFTCAP;
        if (!(diff >= 0 && diff < WIN_SZ)) sv += -1.0e9f;
        ldsS[row][c] = sv;
      }
    }
    __syncthreads();

    if (t < 32) { // per-row running max + rescale factor
      float tm = -3.0e38f;
#pragma unroll
      for (int c = 0; c < 32; ++c) tm = fmaxf(tm, ldsS[t][c]);
      const float nm = fmaxf(rowM[t], tm);
      rowScale[t] = __expf(rowM[t] - nm);
      rowM[t] = nm;
    }
    __syncthreads();

    { // P = exp(S - max) -> bf16; partial row sums
      const int row = t >> 3, c0 = (t & 7) * 4;
      const float nm = rowM[row];
      float ps = 0.f;
#pragma unroll
      for (int e = 0; e < 4; ++e) {
        const float p = __expf(ldsS[row][c0 + e] - nm);
        ldsP[row][c0 + e] = (bf16)p;
        ps += p;
      }
      partS[row][t & 7] = ps;
    }
    __syncthreads();
    if (t < 32) {
      float s = 0.f;
#pragma unroll
      for (int j = 0; j < 8; ++j) s += partS[t][j];
      rowL[t] = rowL[t] * rowScale[t] + s;
    }
    __syncthreads();

    { // rescale O, then O += P @ V (K=32 covers whole key tile); 4 WMMA/wave
      const int mo = (lane >= 16) ? 8 : 0;
#pragma unroll
      for (int r = 0; r < 8; ++r) {
        const float s0f = rowScale[mo + r];
        const float s1f = rowScale[16 + mo + r];
        o00[r] *= s0f; o01[r] *= s0f;
        o10[r] *= s1f; o11[r] *= s1f;
      }
      v16bf a0 = frag_a_rowmajor(&ldsP[lane & 15][0],        lane);
      v16bf a1 = frag_a_rowmajor(&ldsP[16 + (lane & 15)][0], lane);
      const int c0 = wave * 32;
      v16bf b0 = frag_b_kmajor(&ldsV[0][c0],      D_HEAD, lane);
      v16bf b1 = frag_b_kmajor(&ldsV[0][c0 + 16], D_HEAD, lane);
      o00 = __builtin_amdgcn_wmma_f32_16x16x32_bf16(false, a0, false, b0, (short)0, o00, false, false);
      o01 = __builtin_amdgcn_wmma_f32_16x16x32_bf16(false, a0, false, b1, (short)0, o01, false, false);
      o10 = __builtin_amdgcn_wmma_f32_16x16x32_bf16(false, a1, false, b0, (short)0, o10, false, false);
      o11 = __builtin_amdgcn_wmma_f32_16x16x32_bf16(false, a1, false, b1, (short)0, o11, false, false);
    }
    __syncthreads();
  }

  { // finalize: divide by l, write Ctx [s][h*D + d] bf16
    const int mo = (lane >= 16) ? 8 : 0;
    const int n  = lane & 15;
#pragma unroll
    for (int r = 0; r < 8; ++r) {
      const int m0 = mo + r, m1 = 16 + mo + r;
      const float i0 = 1.0f / rowL[m0];
      const float i1 = 1.0f / rowL[m1];
      const size_t b0 = (size_t)(q0 + m0) * (HQ_N * D_HEAD) + (size_t)h * D_HEAD + wave * 32 + n;
      const size_t b1 = (size_t)(q0 + m1) * (HQ_N * D_HEAD) + (size_t)h * D_HEAD + wave * 32 + n;
      Ctx[b0]      = (bf16)(o00[r] * i0);
      Ctx[b0 + 16] = (bf16)(o01[r] * i0);
      Ctx[b1]      = (bf16)(o10[r] * i1);
      Ctx[b1 + 16] = (bf16)(o11[r] * i1);
    }
  }
}

// ================= Kernel 3: output projection out = Ctx @ Wo^T =============
// grid = (S/32, H/256); block = 256. 32x256 tile, 2x2 WMMA subtiles per wave.
__global__ __launch_bounds__(256)
void out_proj_kernel(const bf16* __restrict__ Ctx,
                     const float* __restrict__ Wo,
                     float* __restrict__ out) {
  const int s0   = blockIdx.x * 32;
  const int col0 = blockIdx.y * 256;
  const int t = threadIdx.x, lane = t & 31, wave = t >> 5;

  __shared__ __align__(16) bf16 ldsA[32][32];
  __shared__ __align__(16) bf16 ldsB[256][32];

  v8f acc00 = {}, acc01 = {}, acc10 = {}, acc11 = {};

  for (int k0 = 0; k0 < H_DIM; k0 += 32) {
    { // A: copy 32x32 bf16 (4 per thread, 8B aligned)
      const int r = t >> 3, c = (t & 7) * 4;
      *(uint2*)&ldsA[r][c] =
          *(const uint2*)(Ctx + (size_t)(s0 + r) * H_DIM + k0 + c);
    }
    { // B: 256 Wo rows, f32 -> bf16, n-major
      const float* wr = Wo + (size_t)(col0 + t) * H_DIM + k0;
#pragma unroll
      for (int c = 0; c < 32; c += 4) {
        float4 v = *(const float4*)(wr + c);
        ldsB[t][c]   = (bf16)v.x; ldsB[t][c+1] = (bf16)v.y;
        ldsB[t][c+2] = (bf16)v.z; ldsB[t][c+3] = (bf16)v.w;
      }
    }
    if (k0 + 32 < H_DIM)
      __builtin_prefetch(Wo + (size_t)col0 * H_DIM + k0 + 32, 0, 1);
    __syncthreads();
    {
      v16bf a0 = frag_a_rowmajor(&ldsA[lane & 15][0],        lane);
      v16bf a1 = frag_a_rowmajor(&ldsA[16 + (lane & 15)][0], lane);
      v16bf b0 = frag_b_nmajor(&ldsB[wave * 32][0],      32, lane);
      v16bf b1 = frag_b_nmajor(&ldsB[wave * 32 + 16][0], 32, lane);
      acc00 = __builtin_amdgcn_wmma_f32_16x16x32_bf16(false, a0, false, b0, (short)0, acc00, false, false);
      acc01 = __builtin_amdgcn_wmma_f32_16x16x32_bf16(false, a0, false, b1, (short)0, acc01, false, false);
      acc10 = __builtin_amdgcn_wmma_f32_16x16x32_bf16(false, a1, false, b0, (short)0, acc10, false, false);
      acc11 = __builtin_amdgcn_wmma_f32_16x16x32_bf16(false, a1, false, b1, (short)0, acc11, false, false);
    }
    __syncthreads();
  }

  {
    const int mo = (lane >= 16) ? 8 : 0;
    const int n0 = col0 + wave * 32 + (lane & 15);
#pragma unroll
    for (int r = 0; r < 8; ++r) {
      const int m0 = s0 + mo + r, m1 = s0 + 16 + mo + r;
      out[(size_t)m0 * H_DIM + n0]      = acc00[r];
      out[(size_t)m0 * H_DIM + n0 + 16] = acc01[r];
      out[(size_t)m1 * H_DIM + n0]      = acc10[r];
      out[(size_t)m1 * H_DIM + n0 + 16] = acc11[r];
    }
  }
}

// =========================== launcher ======================================
extern "C" void kernel_launch(void* const* d_in, const int* in_sizes, int n_in,
                              void* d_out, int out_size, void* d_ws, size_t ws_size,
                              hipStream_t stream) {
  (void)in_sizes; (void)n_in; (void)out_size; (void)ws_size;
  const float* hidden = (const float*)d_in[0];
  const float* cosb   = (const float*)d_in[1];
  const float* sinb   = (const float*)d_in[2];
  // d_in[3] = attention_mask: unused, mask computed analytically (sliding causal)
  const float* Wq     = (const float*)d_in[4];
  const float* Wk     = (const float*)d_in[5];
  const float* Wv     = (const float*)d_in[6];
  const float* Wo     = (const float*)d_in[7];
  const float* qnw    = (const float*)d_in[8];
  const float* knw    = (const float*)d_in[9];
  float* out = (float*)d_out;

  // workspace carve-out (bf16): Q 16MB, K 8MB, V 8MB, Ctx 16MB = 48MB
  bf16* Qbuf = (bf16*)d_ws;
  bf16* Kbuf = Qbuf + (size_t)HQ_N  * S_LEN * D_HEAD;
  bf16* Vbuf = Kbuf + (size_t)HKV_N * S_LEN * D_HEAD;
  bf16* CtxB = Vbuf + (size_t)HKV_N * S_LEN * D_HEAD;

  qkv_proj_norm_rope_kernel<<<dim3(S_LEN / 32, HQ_N + 2 * HKV_N), 256, 0, stream>>>(
      hidden, cosb, sinb, Wq, Wk, Wv, qnw, knw, Qbuf, Kbuf, Vbuf);

  flash_attn_kernel<<<dim3(S_LEN / 32, HQ_N), 256, 0, stream>>>(Qbuf, Kbuf, Vbuf, CtxB);

  out_proj_kernel<<<dim3(S_LEN / 32, H_DIM / 256), 256, 0, stream>>>(CtxB, Wo, out);
}
